// GraphMatcherP_89739046683486
// MI455X (gfx1250) — matrix operand order repaired and checked by reading, not compiled, not run on hardware
//
#include <hip/hip_runtime.h>
#include <hip/hip_bf16.h>

typedef __attribute__((ext_vector_type(16))) _Float16 v16h;
typedef __attribute__((ext_vector_type(8)))  float    v8f;

constexpr int NPTS  = 1024;   // keypoints
constexpr int DIM   = 256;    // feature dim
constexpr int BATCH = 4;      // 2*bs
constexpr int HEADS = 4;

union H8 { uint4 u; _Float16 h[8]; };

// ---------------------------------------------------------------------------
// f32 -> f16 conversion (weights)
// ---------------------------------------------------------------------------
__global__ void f32_to_f16_kernel(const float* __restrict__ in,
                                  _Float16* __restrict__ out, long n) {
    long i = (long)blockIdx.x * blockDim.x + threadIdx.x;
    if (i < n) out[i] = (_Float16)in[i];
}

// ---------------------------------------------------------------------------
// build x = concat(desc0, desc1) as f32 residual and f16 (first 256 ch of xh)
// xh layout: [4][512][NPTS], channels 256..511 later filled with "message"
// ---------------------------------------------------------------------------
__global__ void pack_x_kernel(const float* __restrict__ d0,
                              const float* __restrict__ d1,
                              float* __restrict__ x,
                              _Float16* __restrict__ xh) {
    long i = (long)blockIdx.x * blockDim.x + threadIdx.x;
    const long tot = (long)BATCH * DIM * NPTS;
    if (i >= tot) return;
    const long per = (long)2 * DIM * NPTS;
    float v = (i < per) ? d0[i] : d1[i - per];
    x[i] = v;
    long b = i / ((long)DIM * NPTS);
    long rem = i - b * (long)DIM * NPTS;
    xh[b * (long)2 * DIM * NPTS + rem] = (_Float16)v;
}

// ---------------------------------------------------------------------------
// Generic WMMA GEMM:  out[b,m,n] = sum_k W[m,k] * In[in_b,k,n] + bias[m]
//   In: f16, channel stride NPTS, batch stride inBS. in_b = srcSwap?(b+2)&3:b
//   Epilogue: optional +residual (f32), optional f32 store, optional f16 store
//   (scaled by outScale).
// Workgroup: 256 threads = 8 waves as 2(M)x4(N); macro tile 64(M) x 128(N).
// LDS tiles are stored fragment-ready so every lane's WMMA operand is two
// contiguous 16B runs -> ds_load_b128, no v_mov_b16 repacking.
// ---------------------------------------------------------------------------
__global__ __launch_bounds__(256)
void gemm_wmma(const _Float16* __restrict__ W,     // [M,K]
               const float*    __restrict__ bias,  // [M]
               const _Float16* __restrict__ In,
               long inBS, int srcSwap,
               int M, int K,
               float* __restrict__ outF32, long oFBS,
               const float* __restrict__ residual, long resBS,
               _Float16* __restrict__ outF16, long oHBS,
               float outScale) {
    __shared__ __align__(16) _Float16 aT[64][40];    // [m][k]
    __shared__ __align__(16) _Float16 bT[128][40];   // [n][k]  fragment-ready

    const int b    = blockIdx.z;
    const int ib   = srcSwap ? ((b + 2) & 3) : b;
    const int m0   = blockIdx.y * 64;
    const int n0   = blockIdx.x * 128;
    const int tid  = threadIdx.x;
    const int lane = tid & 31;
    const int w    = tid >> 5;
    const int wm   = (w >> 2) * 32;
    const int wn   = (w & 3) * 32;
    const int r    = lane & 15;
    const int hi   = lane >> 4;

    const _Float16* inB = In + (long)ib * inBS;

    // per-thread staging coordinates (8-half = 16B chunks)
    const int amm = tid >> 2,  ack = (tid & 3) * 8;    // A: 64x32 = 256 chunks
    const int bkk0 = tid >> 4, bcn0 = (tid & 15) * 8;  // B: chunk c = tid + g*256

    v8f acc[2][2] = {};

    for (int k0 = 0; k0 < K; k0 += 32) {
        // stage A: vector load + vector store (layout preserved)
        {
            uint4 av = *(const uint4*)&W[(long)(m0 + amm) * K + k0 + ack];
            *(uint4*)&aT[amm][ack] = av;
        }
        // stage B: vector load, transposed scatter to [n][k]
        #pragma unroll
        for (int g = 0; g < 2; g++) {
            int c  = tid + g * 256;     // 0..511
            int kk = c >> 4;            // 0..31
            int cn = (c & 15) * 8;      // 0..120
            H8 t;
            t.u = *(const uint4*)&inB[(long)(k0 + kk) * NPTS + n0 + cn];
            #pragma unroll
            for (int u = 0; u < 8; u++) bT[cn + u][kk] = t.h[u];
        }
        // prefetch next K-step tiles (global_prefetch_b8; L2-resident data)
        if (k0 + 32 < K) {
            __builtin_prefetch(&W[(long)(m0 + amm) * K + k0 + 32 + ack], 0, 0);
            __builtin_prefetch(&inB[(long)(k0 + 32 + bkk0) * NPTS + n0 + bcn0], 0, 0);
        }
        __syncthreads();

        v16h a[2], bb[2];
        #pragma unroll
        for (int i = 0; i < 2; i++)
            #pragma unroll
            for (int e = 0; e < 16; e++) {
                int kk = hi * 8 + e + (e >= 8 ? 8 : 0);
                a[i][e] = aT[wm + i * 16 + r][kk];
            }
        #pragma unroll
        for (int j = 0; j < 2; j++)
            #pragma unroll
            for (int e = 0; e < 16; e++)
                bb[j][e] = bT[wn + j * 16 + r][hi * 8 + e + (e >= 8 ? 8 : 0)];
        #pragma unroll
        for (int i = 0; i < 2; i++)
            #pragma unroll
            for (int j = 0; j < 2; j++)
                acc[i][j] = __builtin_amdgcn_wmma_f32_16x16x32_f16(
                    false, a[i], false, bb[j], (short)0, acc[i][j], false, false);
        __syncthreads();
    }

    // epilogue
    #pragma unroll
    for (int i = 0; i < 2; i++)
        #pragma unroll
        for (int j = 0; j < 2; j++)
            #pragma unroll
            for (int e = 0; e < 8; e++) {
                int m = m0 + wm + i * 16 + hi * 8 + e;
                int n = n0 + wn + j * 16 + r;
                float v = acc[i][j][e] + (bias ? bias[m] : 0.0f);
                if (residual) v += residual[(long)b * resBS + (long)m * NPTS + n];
                if (outF32) outF32[(long)b * oFBS + (long)m * NPTS + n] = v;
                if (outF16) outF16[(long)b * oHBS + (long)m * NPTS + n] =
                    (_Float16)(v * outScale);
            }
}

// ---------------------------------------------------------------------------
// Fused flash attention per (b, h, 64-query block).
// q/k/v f16 [4][256][NPTS], channel c = d*HEADS + h. 1/sqrt(64) pre-folded in q.
// Multiplicative mask (applied before softmax, per reference).
// Block = 128 threads = 4 waves; wave w owns 16 queries.
// ---------------------------------------------------------------------------
__global__ __launch_bounds__(128)
void attn_wmma(const _Float16* __restrict__ qh,
               const _Float16* __restrict__ kh,
               const _Float16* __restrict__ vh,
               const float* __restrict__ M0p,
               const float* __restrict__ M1p,
               _Float16* __restrict__ attnh) {
    __shared__ __align__(16) _Float16 kT[32][72];      // [m][d] fragment-ready
    __shared__ __align__(16) _Float16 vT[64][40];      // [d][m] fragment-ready
    __shared__ __align__(16) _Float16 pT[4][16][40];   // per-wave P tile 16x32

    const int b    = blockIdx.z;
    const int h    = blockIdx.y;
    const int tid  = threadIdx.x;
    const int lane = tid & 31;
    const int w    = tid >> 5;
    const int r    = lane & 15;
    const int hi   = lane >> 4;
    const int q0   = blockIdx.x * 64 + w * 16;

    const long chBase = (long)b * DIM * NPTS;
    const float* mask = (b < 2) ? (M0p + (long)b * NPTS * NPTS)
                                : (M1p + (long)(b - 2) * NPTS * NPTS);

    // Q fragments (A layout), K-dim = d = 64 -> 2 fragments
    v16h qa[2];
    #pragma unroll
    for (int t = 0; t < 2; t++)
        #pragma unroll
        for (int e = 0; e < 16; e++) {
            int kk = hi * 8 + e + (e >= 8 ? 8 : 0);
            int d = t * 32 + kk;
            qa[t][e] = qh[chBase + (long)(d * HEADS + h) * NPTS + q0 + r];
        }

    // staging coords: K/V tile = 64 ch x 32 m = 256 chunks of 8, 2 per thread
    float rm[8], rl[8];
    v8f o[4] = {};
    #pragma unroll
    for (int e = 0; e < 8; e++) { rm[e] = -1e30f; rl[e] = 0.0f; }

    for (int mB = 0; mB < NPTS; mB += 32) {
        __syncthreads();   // protect previous iteration's kT/vT reads
        #pragma unroll
        for (int g = 0; g < 2; g++) {
            int c  = tid + g * 128;   // 0..255
            int d  = c >> 2;          // 0..63
            int cm = (c & 3) * 8;     // 0..24
            const long src = chBase + (long)(d * HEADS + h) * NPTS + mB + cm;
            H8 tk; tk.u = *(const uint4*)&kh[src];
            #pragma unroll
            for (int u = 0; u < 8; u++) kT[cm + u][d] = tk.h[u];  // transpose
            uint4 tv = *(const uint4*)&vh[src];
            *(uint4*)&vT[d][cm] = tv;                              // direct
        }
        if (mB + 32 < NPTS) {
            const long nxt = chBase + (long)((tid >> 2) * HEADS + h) * NPTS
                           + mB + 32 + (tid & 3) * 8;
            __builtin_prefetch(&kh[nxt], 0, 0);
            __builtin_prefetch(&vh[nxt], 0, 0);
        }
        __syncthreads();

        // scores S[16q x 32m], K = 64 (two chained WMMAs per 16-col tile)
        v8f s[2] = {};
        #pragma unroll
        for (int j = 0; j < 2; j++)
            #pragma unroll
            for (int t = 0; t < 2; t++) {
                v16h kb;
                #pragma unroll
                for (int e = 0; e < 16; e++)
                    kb[e] = kT[j * 16 + r][t * 32 + hi * 8 + e + (e >= 8 ? 8 : 0)];
                s[j] = __builtin_amdgcn_wmma_f32_16x16x32_f16(
                    false, qa[t], false, kb, (short)0, s[j], false, false);
            }

        // multiplicative mask
        #pragma unroll
        for (int j = 0; j < 2; j++)
            #pragma unroll
            for (int e = 0; e < 8; e++) {
                int qq = q0 + hi * 8 + e;
                int mm = mB + j * 16 + r;
                s[j][e] *= mask[(long)qq * NPTS + mm];
            }

        // online softmax (per-row state replicated across the 16-lane group)
        #pragma unroll
        for (int e = 0; e < 8; e++) {
            float le = fmaxf(s[0][e], s[1][e]);
            #pragma unroll
            for (int off = 1; off < 16; off <<= 1)
                le = fmaxf(le, __shfl_xor(le, off, 32));
            float nm = fmaxf(rm[e], le);
            float sc = __expf(rm[e] - nm);
            float e0 = __expf(s[0][e] - nm);
            float e1 = __expf(s[1][e] - nm);
            float rs = e0 + e1;
            #pragma unroll
            for (int off = 1; off < 16; off <<= 1)
                rs += __shfl_xor(rs, off, 32);
            rl[e] = rl[e] * sc + rs;
            rm[e] = nm;
            s[0][e] = e0; s[1][e] = e1;
            #pragma unroll
            for (int t2 = 0; t2 < 4; t2++) o[t2][e] *= sc;
        }

        // P (C layout f32) -> LDS -> A layout f16 (per-wave buffer: DScnt
        // drain is sufficient, no workgroup barrier needed)
        #pragma unroll
        for (int j = 0; j < 2; j++)
            #pragma unroll
            for (int e = 0; e < 8; e++)
                pT[w][hi * 8 + e][j * 16 + r] = (_Float16)s[j][e];
        asm volatile("s_wait_dscnt 0x0" ::: "memory");

        v16h pa;
        #pragma unroll
        for (int e = 0; e < 16; e++) {
            int kk = hi * 8 + e + (e >= 8 ? 8 : 0);
            pa[e] = pT[w][r][kk];
        }
        // O[16q x 64d] += P @ V
        #pragma unroll
        for (int t2 = 0; t2 < 4; t2++) {
            v16h vb;
            #pragma unroll
            for (int e = 0; e < 16; e++)
                vb[e] = vT[t2 * 16 + r][hi * 8 + e + (e >= 8 ? 8 : 0)];
            o[t2] = __builtin_amdgcn_wmma_f32_16x16x32_f16(
                false, pa, false, vb, (short)0, o[t2], false, false);
        }
    }

    // normalize and emit f16 at channel c = d*HEADS + h
    #pragma unroll
    for (int t2 = 0; t2 < 4; t2++)
        #pragma unroll
        for (int e = 0; e < 8; e++) {
            int qq = q0 + hi * 8 + e;
            int d = t2 * 16 + r;
            float val = o[t2][e] / rl[e];
            attnh[chBase + (long)(d * HEADS + h) * NPTS + qq] = (_Float16)val;
        }
}

// ---------------------------------------------------------------------------
// Training-mode BatchNorm1d over (B, N) per channel + ReLU, f16 output.
// One workgroup per channel (4096 samples). Population variance, eps=1e-5.
// ---------------------------------------------------------------------------
__global__ __launch_bounds__(256)
void bn_relu_kernel(const float* __restrict__ z,
                    const float* __restrict__ g,
                    const float* __restrict__ beta,
                    _Float16* __restrict__ yh, int C) {
    __shared__ float s1[256], s2[256];
    const int c = blockIdx.x;
    const int tid = threadIdx.x;
    const int total = BATCH * NPTS;
    float sum = 0.f, sq = 0.f;
    for (int t = tid; t < total; t += 256) {
        int bb = t >> 10, nn = t & (NPTS - 1);
        float v = z[((long)bb * C + c) * NPTS + nn];
        sum += v; sq += v * v;
    }
    s1[tid] = sum; s2[tid] = sq;
    __syncthreads();
    for (int st = 128; st > 0; st >>= 1) {
        if (tid < st) { s1[tid] += s1[tid + st]; s2[tid] += s2[tid + st]; }
        __syncthreads();
    }
    float mean = s1[0] / total;
    float var  = s2[0] / total - mean * mean;
    float inv  = rsqrtf(var + 1e-5f);
    float gg = g[c], bt = beta[c];
    for (int t = tid; t < total; t += 256) {
        int bb = t >> 10, nn = t & (NPTS - 1);
        long idx = ((long)bb * C + c) * NPTS + nn;
        float v = (z[idx] - mean) * inv * gg + bt;
        yh[idx] = (_Float16)fmaxf(v, 0.0f);
    }
}

// ---------------------------------------------------------------------------
// host driver
// ---------------------------------------------------------------------------
extern "C" void kernel_launch(void* const* d_in, const int* in_sizes, int n_in,
                              void* d_out, int out_size, void* d_ws, size_t ws_size,
                              hipStream_t stream) {
    const float* desc0 = (const float*)d_in[0];
    const float* desc1 = (const float*)d_in[1];
    const float* M0p   = (const float*)d_in[2];
    const float* M1p   = (const float*)d_in[3];
    const float* Wq = (const float*)d_in[4];
    const float* bq = (const float*)d_in[5];
    const float* Wk = (const float*)d_in[6];
    const float* bk = (const float*)d_in[7];
    const float* Wv = (const float*)d_in[8];
    const float* bv = (const float*)d_in[9];
    const float* Wm = (const float*)d_in[10];
    const float* bm = (const float*)d_in[11];
    const float* W1 = (const float*)d_in[12];
    const float* b1 = (const float*)d_in[13];
    const float* g1 = (const float*)d_in[14];
    const float* be1 = (const float*)d_in[15];
    const float* W2 = (const float*)d_in[16];
    const float* b2 = (const float*)d_in[17];

    // workspace carve
    char* p = (char*)d_ws;
    auto carve = [&](size_t bytes) -> char* {
        char* r = p; p += (bytes + 255) & ~(size_t)255; return r;
    };
    float*    x   = (float*)   carve((size_t)BATCH * DIM * NPTS * 4);
    _Float16* xh  = (_Float16*)carve((size_t)BATCH * 2 * DIM * NPTS * 2);
    _Float16* qhb = (_Float16*)carve((size_t)BATCH * DIM * NPTS * 2);
    _Float16* khb = (_Float16*)carve((size_t)BATCH * DIM * NPTS * 2);
    _Float16* vhb = (_Float16*)carve((size_t)BATCH * DIM * NPTS * 2);
    _Float16* ahb = (_Float16*)carve((size_t)BATCH * DIM * NPTS * 2);
    float*    z   = (float*)   carve((size_t)BATCH * 2 * DIM * NPTS * 4);
    _Float16* yh  = (_Float16*)carve((size_t)BATCH * 2 * DIM * NPTS * 2);
    _Float16* Wqh = (_Float16*)carve((size_t)6 * DIM * DIM * 2);
    _Float16* Wkh = (_Float16*)carve((size_t)6 * DIM * DIM * 2);
    _Float16* Wvh = (_Float16*)carve((size_t)6 * DIM * DIM * 2);
    _Float16* Wmh = (_Float16*)carve((size_t)6 * DIM * DIM * 2);
    _Float16* W1h = (_Float16*)carve((size_t)6 * 2 * DIM * 2 * DIM * 2);
    _Float16* W2h = (_Float16*)carve((size_t)6 * DIM * 2 * DIM * 2);

    auto conv = [&](const float* s, _Float16* d, long n) {
        f32_to_f16_kernel<<<dim3((unsigned)((n + 255) / 256)), dim3(256), 0, stream>>>(s, d, n);
    };
    conv(Wq, Wqh, 6L * DIM * DIM);
    conv(Wk, Wkh, 6L * DIM * DIM);
    conv(Wv, Wvh, 6L * DIM * DIM);
    conv(Wm, Wmh, 6L * DIM * DIM);
    conv(W1, W1h, 6L * 2 * DIM * 2 * DIM);
    conv(W2, W2h, 6L * DIM * 2 * DIM);

    {
        long tot = (long)BATCH * DIM * NPTS;
        pack_x_kernel<<<dim3((unsigned)((tot + 255) / 256)), dim3(256), 0, stream>>>(
            desc0, desc1, x, xh);
    }

    const long bsX  = (long)2 * DIM * NPTS;   // xh / z / yh batch stride (512 ch)
    const long bsD  = (long)DIM * NPTS;       // 256-channel batch stride
    dim3 blk(256);
    dim3 gProj(NPTS / 128, DIM / 64, BATCH);       // M=256
    dim3 gMlp1(NPTS / 128, (2 * DIM) / 64, BATCH); // M=512
    dim3 gAttn(NPTS / 64, HEADS, BATCH);

    for (int i = 0; i < 6; i++) {
        int cross = i & 1;  // self,cross,self,cross,self,cross
        // Q/K/V projections (q gets 1/sqrt(head_dim) folded in)
        gemm_wmma<<<gProj, blk, 0, stream>>>(Wqh + (long)i * DIM * DIM, bq + i * DIM,
            xh, bsX, 0, DIM, DIM, nullptr, 0, nullptr, 0, qhb, bsD, 0.125f);
        gemm_wmma<<<gProj, blk, 0, stream>>>(Wkh + (long)i * DIM * DIM, bk + i * DIM,
            xh, bsX, cross, DIM, DIM, nullptr, 0, nullptr, 0, khb, bsD, 1.0f);
        gemm_wmma<<<gProj, blk, 0, stream>>>(Wvh + (long)i * DIM * DIM, bv + i * DIM,
            xh, bsX, cross, DIM, DIM, nullptr, 0, nullptr, 0, vhb, bsD, 1.0f);
        // fused masked attention
        attn_wmma<<<gAttn, dim3(128), 0, stream>>>(qhb, khb, vhb, M0p, M1p, ahb);
        // merge -> message (f16) into channels 256..511 of xh
        gemm_wmma<<<gProj, blk, 0, stream>>>(Wmh + (long)i * DIM * DIM, bm + i * DIM,
            ahb, bsD, 0, DIM, DIM, nullptr, 0, nullptr, 0,
            xh + (long)DIM * NPTS, bsX, 1.0f);
        // MLP1: z = W1 @ [x; message] + b1   (f32 pre-BN)
        gemm_wmma<<<gMlp1, blk, 0, stream>>>(W1h + (long)i * 2 * DIM * 2 * DIM,
            b1 + i * 2 * DIM, xh, bsX, 0, 2 * DIM, 2 * DIM,
            z, bsX, nullptr, 0, nullptr, 0, 1.0f);
        // BatchNorm + ReLU -> yh (f16)
        bn_relu_kernel<<<dim3(2 * DIM), dim3(256), 0, stream>>>(
            z, g1 + i * 2 * DIM, be1 + i * 2 * DIM, yh, 2 * DIM);
        // MLP2 + residual: x' = x + W2 @ yh + b2 ; last layer writes d_out
        float* oz = (i == 5) ? (float*)d_out : x;
        gemm_wmma<<<gProj, blk, 0, stream>>>(W2h + (long)i * DIM * 2 * DIM,
            b2 + i * DIM, yh, bsX, 0, DIM, 2 * DIM,
            oz, bsD, x, bsD, xh, bsX, 1.0f);
    }
}
